// SelfAttentionLayer_61916248539687
// MI455X (gfx1250) — compile-verified, matrix operand order checked by reading, not run on hardware
//
#include <hip/hip_runtime.h>

#define BB 2
#define SS 2048
#define DD 1024
#define HH 16
#define HDM 64
#define MTOT (BB * SS)  // 4096

typedef __attribute__((ext_vector_type(16))) _Float16 v16h;
typedef __attribute__((ext_vector_type(8)))  _Float16 v8h;
typedef __attribute__((ext_vector_type(4)))  _Float16 v4h;
typedef __attribute__((ext_vector_type(8)))  float    v8f;

// Load a 16x32 f16 A-style fragment (also used for B with lane = column),
// per CDNA5 ISA 7.12.2: lane half hi=0 holds K {0..7,16..23}, hi=1 holds
// K {8..15,24..31}.  p = row start (+ k_base); 16B-aligned rows.
__device__ __forceinline__ v16h load_frag(const _Float16* p, int hi) {
  v8h lo = *(const v8h*)(p + hi * 8);
  v8h hh = *(const v8h*)(p + 16 + hi * 8);
  v16h f;
#pragma unroll
  for (int i = 0; i < 8; ++i) { f[i] = lo[i]; f[i + 8] = hh[i]; }
  return f;
}

__device__ __forceinline__ v8f wmma_f16(v16h a, v16h b, v8f c) {
  return __builtin_amdgcn_wmma_f32_16x16x32_f16(false, a, false, b, (short)0, c,
                                                false, false);
}

// CDNA5 async copy global -> LDS, 16B per lane, tracked by ASYNCcnt.
__device__ __forceinline__ void async_ld_b128(_Float16* lds, const _Float16* g) {
  uint32_t loff = (uint32_t)(uintptr_t)lds;  // LDS aperture: low 32 bits = offset
  asm volatile("global_load_async_to_lds_b128 %0, %1, off"
               :: "v"(loff), "v"((uint64_t)(uintptr_t)g)
               : "memory");
}
__device__ __forceinline__ void wait_async0() {
  asm volatile("s_wait_asynccnt 0" ::: "memory");
}

// ---- Kernel 1: convert + transpose weights: Wt[w][n][k] = (f16) W[w][k][n]
__global__ void sa_cvt_weights(const float* __restrict__ Wq,
                               const float* __restrict__ Wk,
                               const float* __restrict__ Wv,
                               const float* __restrict__ Wo,
                               _Float16* __restrict__ Wt) {
  int idx = blockIdx.x * blockDim.x + threadIdx.x;  // 0 .. 4*D*D-1
  int w = idx >> 20;
  int rem = idx & (DD * DD - 1);
  int k = rem >> 10;
  int n = rem & (DD - 1);
  const float* src = (w == 0) ? Wq : (w == 1) ? Wk : (w == 2) ? Wv : Wo;
  Wt[(size_t)w * DD * DD + (size_t)n * DD + k] = (_Float16)src[(size_t)k * DD + n];
}

// ---- Kernel 2: QKV projection GEMM (pipelined). z=0:Q(*1/8), 1:K, 2:V(transposed)
__global__ __launch_bounds__(256) void sa_qkv_gemm(
    const float* __restrict__ Xq, const float* __restrict__ Xk,
    const float* __restrict__ Xv, const _Float16* __restrict__ Wt,
    const float* __restrict__ bq, const float* __restrict__ bk,
    const float* __restrict__ bv, _Float16* __restrict__ Q,
    _Float16* __restrict__ K, _Float16* __restrict__ Vt) {
  __shared__ _Float16 smem[16384];  // [A0|A1|B0|B1] 4096 halves each; epi: 128x128
  const int z = blockIdx.z;
  const float* X = (z == 0) ? Xq : (z == 1) ? Xk : Xv;
  const float* bias = (z == 0) ? bq : (z == 1) ? bk : bv;
  const _Float16* W = Wt + (size_t)z * DD * DD;  // [n][k]
  const int m0 = blockIdx.x * 128;
  const int n0 = blockIdx.y * 128;
  const int tid = threadIdx.x;
  const int lane = tid & 31, wave = tid >> 5;
  const int ln = lane & 15, hi = lane >> 4;
  const int wm = wave >> 1, wn = wave & 1;  // 4x2 waves -> 32x64 per wave

  v8f acc[2][4] = {};
  float4 ra[4];

  // buffer bases computed arithmetically (no LDS-pointer arrays!)
  auto bufA = [&](int bf) -> _Float16* { return smem + bf * 4096; };
  auto bufB = [&](int bf) -> _Float16* { return smem + 8192 + bf * 4096; };

  auto load_ra = [&](int k0) {
#pragma unroll
    for (int i = 0; i < 4; ++i) {
      int u = tid + i * 256, row = u >> 3, c4 = (u & 7) * 4;
      ra[i] = *(const float4*)(X + (size_t)(m0 + row) * DD + k0 + c4);
    }
  };
  auto stage_A = [&](_Float16* dst) {  // convert regs -> LDS
#pragma unroll
    for (int i = 0; i < 4; ++i) {
      int u = tid + i * 256, row = u >> 3, c4 = (u & 7) * 4;
      v4h h4;
      h4[0] = (_Float16)ra[i].x; h4[1] = (_Float16)ra[i].y;
      h4[2] = (_Float16)ra[i].z; h4[3] = (_Float16)ra[i].w;
      *(v4h*)(dst + row * 32 + c4) = h4;
    }
  };
  auto async_B = [&](_Float16* dst, int k0) {
#pragma unroll
    for (int i = 0; i < 2; ++i) {
      int u = tid + i * 256, row = u >> 2, c = u & 3;  // dst off = u*8 halves
      async_ld_b128(dst + u * 8, W + (size_t)(n0 + row) * DD + k0 + c * 8);
    }
  };

  // prologue: tile 0 staged, tile 1's A prefetched to regs
  load_ra(0);
  stage_A(bufA(0));
  async_B(bufB(0), 0);
  load_ra(32);
  wait_async0();
  __syncthreads();

  int buf = 0;
  for (int kk = 0; kk < 32; ++kk) {
    if (kk < 31) {                       // stage tile kk+1 into other buffer
      stage_A(bufA(buf ^ 1));            // consumes ra(kk+1)
      async_B(bufB(buf ^ 1), (kk + 1) * 32);
      if (kk < 30) load_ra((kk + 2) * 32);
    }
    // compute tile kk
    const _Float16* cA = bufA(buf);
    const _Float16* cB = bufB(buf);
    v16h a[2];
#pragma unroll
    for (int am = 0; am < 2; ++am)
      a[am] = load_frag(cA + (wm * 32 + am * 16 + ln) * 32, hi);
#pragma unroll
    for (int bn = 0; bn < 4; ++bn) {
      v16h b = load_frag(cB + (wn * 64 + bn * 16 + ln) * 32, hi);
#pragma unroll
      for (int am = 0; am < 2; ++am) acc[am][bn] = wmma_f16(a[am], b, acc[am][bn]);
    }
    if (kk < 31) { wait_async0(); __syncthreads(); }
    buf ^= 1;
  }

  // epilogue: stage block to LDS ([m][n], or [n][m] for z==2), packed b128 out
  const float qscale = (z == 0) ? 0.125f : 1.0f;  // 1/sqrt(HD) folded into Q
  __syncthreads();
#pragma unroll
  for (int am = 0; am < 2; ++am) {
#pragma unroll
    for (int bn = 0; bn < 4; ++bn) {
      int n = wn * 64 + bn * 16 + ln;          // C layout: N = lane&15
      float bv_ = bias[n0 + n];
#pragma unroll
      for (int r = 0; r < 8; ++r) {            // M = 8*hi + r
        int m = wm * 32 + am * 16 + hi * 8 + r;
        _Float16 val = (_Float16)((acc[am][bn][r] + bv_) * qscale);
        if (z == 2) smem[n * 128 + m] = val;
        else        smem[m * 128 + n] = val;
      }
    }
  }
  __syncthreads();
#pragma unroll
  for (int i = 0; i < 8; ++i) {
    int u = tid + i * 256;                     // 2048 v8h units, 16 per row
    int rrow = u >> 4, cc8 = (u & 15) * 8;
    v8h val = *(v8h*)(smem + rrow * 128 + cc8);
    if (z == 2) {  // smem is [n][m]: 8 consecutive s
      int n_g = n0 + rrow, h = n_g >> 6, hd = n_g & 63;
      int m_g = m0 + cc8, bi = m_g >> 11, s = m_g & (SS - 1);
      *(v8h*)(Vt + (((size_t)bi * HH + h) * HDM + hd) * SS + s) = val;
    } else {       // smem is [m][n]: 8 consecutive hd
      int m_g = m0 + rrow, bi = m_g >> 11, s = m_g & (SS - 1);
      int n_g = n0 + cc8, h = n_g >> 6, hd = n_g & 63;
      _Float16* dst = (z == 0) ? Q : K;
      *(v8h*)(dst + (((size_t)bi * HH + h) * SS + s) * HDM + hd) = val;
    }
  }
}

// ---- Kernel 3: flash attention per (b,h, 128-row q tile), K/V double-buffered
__global__ __launch_bounds__(256) void sa_attention(
    const _Float16* __restrict__ Q, const _Float16* __restrict__ K,
    const _Float16* __restrict__ Vt, _Float16* __restrict__ attn) {
  __shared__ _Float16 Ksm[2][128 * 64];      // key tile [key][hd]
  __shared__ _Float16 Vsm[2][64 * 128];      // value tile transposed [hd][key]
  __shared__ _Float16 Psm[8 * 16 * 128];     // per-wave P staging [16][128]
  const int qt = blockIdx.x;
  const int bh = blockIdx.y;
  const int b = bh >> 4, h = bh & 15;
  const int tid = threadIdx.x;
  const int lane = tid & 31, wave = tid >> 5;
  const int ln = lane & 15, hi = lane >> 4;
  const int q0 = qt * 128;
  const _Float16* Qh = Q + ((size_t)bh * SS) * HDM;
  const _Float16* Kh = K + ((size_t)bh * SS) * HDM;
  const _Float16* Vh = Vt + ((size_t)bh * HDM) * SS;

  auto stage_kv = [&](int bf, int kt) {      // async fill of one K/V tile pair
    const int key0 = kt * 128;
#pragma unroll
    for (int i = 0; i < 4; ++i) {            // K: contiguous 16KB
      int u = tid + i * 256;
      async_ld_b128(&Ksm[bf][u * 8], Kh + (size_t)key0 * HDM + u * 8);
    }
#pragma unroll
    for (int i = 0; i < 4; ++i) {            // V: 64 rows of 256B from [HD][S]
      int u = tid + i * 256, row = u >> 4, c = u & 15;
      async_ld_b128(&Vsm[bf][u * 8], Vh + (size_t)row * SS + key0 + c * 8);
    }
  };

  v16h aq[2];  // this wave's 16 Q rows, K = 0..31 / 32..63
#pragma unroll
  for (int ks = 0; ks < 2; ++ks)
    aq[ks] = load_frag(Qh + (size_t)(q0 + wave * 16 + ln) * HDM + ks * 32, hi);

  v8f o[4] = {};
  float l[8], mrow[8];
#pragma unroll
  for (int r = 0; r < 8; ++r) { l[r] = 0.f; mrow[r] = -3.0e38f; }

  _Float16* Pw = Psm + wave * (16 * 128);

  stage_kv(0, 0);
  wait_async0();
  __syncthreads();

  int buf = 0;
  for (int kt = 0; kt < SS / 128; ++kt) {
    if (kt < SS / 128 - 1) stage_kv(buf ^ 1, kt + 1);  // prefetch next tile

    // scores: 16 rows x 128 keys, f32 accum
    v8f sc[8];
#pragma unroll
    for (int nt = 0; nt < 8; ++nt) {
      v8f s = {};
#pragma unroll
      for (int ks = 0; ks < 2; ++ks) {
        v16h bfr = load_frag(&Ksm[buf][(nt * 16 + ln) * 64 + ks * 32], hi);
        s = wmma_f16(aq[ks], bfr, s);
      }
      sc[nt] = s;
    }
    // online softmax: rows 8*hi+r live in one 16-lane half -> xor masks 1,2,4,8
    float mnew[8];
#pragma unroll
    for (int r = 0; r < 8; ++r) {
      float v = sc[0][r];
#pragma unroll
      for (int nt = 1; nt < 8; ++nt) v = fmaxf(v, sc[nt][r]);
#pragma unroll
      for (int msk = 1; msk < 16; msk <<= 1) v = fmaxf(v, __shfl_xor(v, msk, 32));
      mnew[r] = fmaxf(v, mrow[r]);
    }
    float scale[8], ps[8];
#pragma unroll
    for (int r = 0; r < 8; ++r) {
      scale[r] = __expf(mrow[r] - mnew[r]);
      mrow[r] = mnew[r];
      ps[r] = 0.f;
    }
#pragma unroll
    for (int nt = 0; nt < 8; ++nt) {
#pragma unroll
      for (int r = 0; r < 8; ++r) {
        float p = __expf(sc[nt][r] - mrow[r]);
        sc[nt][r] = p;
        ps[r] += p;
      }
    }
#pragma unroll
    for (int r = 0; r < 8; ++r) {
#pragma unroll
      for (int msk = 1; msk < 16; msk <<= 1) ps[r] += __shfl_xor(ps[r], msk, 32);
      l[r] = l[r] * scale[r] + ps[r];
    }
#pragma unroll
    for (int t = 0; t < 4; ++t) {
#pragma unroll
      for (int r = 0; r < 8; ++r) o[t][r] *= scale[r];
    }
    // P -> wave-private LDS [16][128] (per-wave LDS ops stay in order)
#pragma unroll
    for (int nt = 0; nt < 8; ++nt) {
#pragma unroll
      for (int r = 0; r < 8; ++r)
        Pw[(hi * 8 + r) * 128 + nt * 16 + ln] = (_Float16)sc[nt][r];
    }
    // O += P @ V   (K dim = 128 keys, 4 wmma k-steps)
#pragma unroll
    for (int ks = 0; ks < 4; ++ks) {
      v16h pa = load_frag(Pw + ln * 128 + ks * 32, hi);
#pragma unroll
      for (int t = 0; t < 4; ++t) {
        v16h vb = load_frag(&Vsm[buf][(t * 16 + ln) * 128 + ks * 32], hi);
        o[t] = wmma_f16(pa, vb, o[t]);
      }
    }
    if (kt < SS / 128 - 1) { wait_async0(); __syncthreads(); }
    buf ^= 1;
  }
  // normalize and write attn[b][s][h*HD + hd]
  float inv[8];
#pragma unroll
  for (int r = 0; r < 8; ++r) inv[r] = 1.0f / l[r];
#pragma unroll
  for (int t = 0; t < 4; ++t) {
#pragma unroll
    for (int r = 0; r < 8; ++r) {
      int s = q0 + wave * 16 + hi * 8 + r;
      int col = h * HDM + t * 16 + ln;
      attn[((size_t)b * SS + s) * DD + col] = (_Float16)(o[t][r] * inv[r]);
    }
  }
}

// ---- Kernel 4: output projection (pipelined, fully async staging), fp32 out
__global__ __launch_bounds__(256) void sa_out_gemm(
    const _Float16* __restrict__ A, const _Float16* __restrict__ Wot,
    const float* __restrict__ bo, float* __restrict__ out) {
  __shared__ _Float16 smem[16384];  // [A0|A1|B0|B1] 4096 halves each
  const int m0 = blockIdx.x * 128, n0 = blockIdx.y * 128;
  const int tid = threadIdx.x;
  const int lane = tid & 31, wave = tid >> 5;
  const int ln = lane & 15, hi = lane >> 4;
  const int wm = wave >> 1, wn = wave & 1;

  auto bufA = [&](int bf) -> _Float16* { return smem + bf * 4096; };
  auto bufB = [&](int bf) -> _Float16* { return smem + 8192 + bf * 4096; };

  auto stage = [&](int bf, int k0) {
    _Float16* dA = bufA(bf);
    _Float16* dB = bufB(bf);
#pragma unroll
    for (int i = 0; i < 2; ++i) {
      int u = tid + i * 256, row = u >> 2, c = u & 3;
      async_ld_b128(dA + u * 8, A   + (size_t)(m0 + row) * DD + k0 + c * 8);
      async_ld_b128(dB + u * 8, Wot + (size_t)(n0 + row) * DD + k0 + c * 8);
    }
  };

  v8f acc[2][4] = {};

  stage(0, 0);
  wait_async0();
  __syncthreads();

  int buf = 0;
  for (int kk = 0; kk < 32; ++kk) {
    if (kk < 31) stage(buf ^ 1, (kk + 1) * 32);
    const _Float16* cA = bufA(buf);
    const _Float16* cB = bufB(buf);
    v16h a[2];
#pragma unroll
    for (int am = 0; am < 2; ++am)
      a[am] = load_frag(cA + (wm * 32 + am * 16 + ln) * 32, hi);
#pragma unroll
    for (int bn = 0; bn < 4; ++bn) {
      v16h b = load_frag(cB + (wn * 64 + bn * 16 + ln) * 32, hi);
#pragma unroll
      for (int am = 0; am < 2; ++am) acc[am][bn] = wmma_f16(a[am], b, acc[am][bn]);
    }
    if (kk < 31) { wait_async0(); __syncthreads(); }
    buf ^= 1;
  }
#pragma unroll
  for (int am = 0; am < 2; ++am) {
#pragma unroll
    for (int bn = 0; bn < 4; ++bn) {
      int n_g = n0 + wn * 64 + bn * 16 + ln;
      float bv_ = bo[n_g];
#pragma unroll
      for (int r = 0; r < 8; ++r) {
        int m_g = m0 + wm * 32 + am * 16 + hi * 8 + r;
        out[(size_t)m_g * DD + n_g] = acc[am][bn][r] + bv_;
      }
    }
  }
}

extern "C" void kernel_launch(void* const* d_in, const int* in_sizes, int n_in,
                              void* d_out, int out_size, void* d_ws,
                              size_t ws_size, hipStream_t stream) {
  (void)in_sizes; (void)n_in; (void)out_size; (void)ws_size;
  const float* q_in = (const float*)d_in[0];
  const float* k_in = (const float*)d_in[1];
  const float* v_in = (const float*)d_in[2];
  const float* Wq = (const float*)d_in[3];
  const float* bq = (const float*)d_in[4];
  const float* Wk = (const float*)d_in[5];
  const float* bk = (const float*)d_in[6];
  const float* Wv = (const float*)d_in[7];
  const float* bv = (const float*)d_in[8];
  const float* Wo = (const float*)d_in[9];
  const float* bo = (const float*)d_in[10];
  float* out = (float*)d_out;

  char* ws = (char*)d_ws;
  const size_t MB = 1024 * 1024;
  _Float16* Wt   = (_Float16*)(ws);             // 4 * D*D f16 = 8MB, [w][n][k]
  _Float16* Qh   = (_Float16*)(ws + 8 * MB);    // [B,H,S,HD] f16, 8MB
  _Float16* Kh   = (_Float16*)(ws + 16 * MB);   // [B,H,S,HD] f16, 8MB
  _Float16* Vth  = (_Float16*)(ws + 24 * MB);   // [B,H,HD,S] f16, 8MB
  _Float16* attn = (_Float16*)(ws + 32 * MB);   // [B,S,D]   f16, 8MB

  sa_cvt_weights<<<dim3((4 * DD * DD) / 256), dim3(256), 0, stream>>>(
      Wq, Wk, Wv, Wo, Wt);
  sa_qkv_gemm<<<dim3(MTOT / 128, DD / 128, 3), dim3(256), 0, stream>>>(
      q_in, k_in, v_in, Wt, bq, bk, bv, Qh, Kh, Vth);
  sa_attention<<<dim3(SS / 128, BB * HH), dim3(256), 0, stream>>>(
      Qh, Kh, Vth, attn);
  sa_out_gemm<<<dim3(MTOT / 128, DD / 128), dim3(256), 0, stream>>>(
      attn, Wt + (size_t)3 * DD * DD, bo, out);
}
